// PositionalAttention_48988396978422
// MI455X (gfx1250) — matrix operand order repaired and checked by reading, not compiled, not run on hardware
//
#include <hip/hip_runtime.h>
#include <hip/hip_bf16.h>

typedef __attribute__((ext_vector_type(16))) _Float16 v16h;
typedef __attribute__((ext_vector_type(8)))  float    v8f;

namespace {
constexpr int kB  = 4;
constexpr int kC  = 512;
constexpr int kCQ = 64;
constexpr int kN  = 4096;          // 64*64
constexpr int kD  = kC + 2 * kCQ;  // 640 fused projection rows: [Wq; Wk; Wv]

// workspace layout (bytes)
constexpr size_t WH_OFF   = 0;                                   // f16 [640][512]
constexpr size_t WH_SZ    = (size_t)kD * kC * 2;
constexpr size_t BIAS_OFF = WH_OFF + WH_SZ;                      // f32 [640]
constexpr size_t BIAS_SZ  = (size_t)kD * 4;
constexpr size_t XH_OFF   = (BIAS_OFF + BIAS_SZ + 255) & ~(size_t)255; // f16 [B][C][N]
constexpr size_t XH_SZ    = (size_t)kB * kC * kN * 2;
constexpr size_t QT_OFF   = XH_OFF + XH_SZ;                      // f16 [B][N][64]  (Q transposed)
constexpr size_t QT_SZ    = (size_t)kB * kN * kCQ * 2;
constexpr size_t KT_OFF   = QT_OFF + QT_SZ;                      // f16 [B][N][64]  (K transposed)
constexpr size_t VH_OFF   = KT_OFF + QT_SZ;                      // f16 [B][C][N]
}

// ---------- WMMA helpers (CDNA5 v_wmma_f32_16x16x32_f16) ----------------
__device__ __forceinline__ v8f wmma_f16(v16h a, v16h b, v8f c) {
  return __builtin_amdgcn_wmma_f32_16x16x32_f16(false, a, false, b, (short)0, c,
                                                false, false);
}

// A tile 16x32 f16, element (m,k) at base[m*stride + k].
// Lanes 0-15: row m=lane, K 0..7 in a[0..7], K 16..23 in a[8..15].
// Lanes 16-31: row m=lane-16, K 8..15 / 24..31.
__device__ __forceinline__ v16h ld_a(const _Float16* base, int stride) {
  int lane = threadIdx.x & 31;
  int row  = lane & 15;
  int kb   = (lane >> 4) << 3;  // 0 or 8
  const _Float16* p = base + row * stride + kb;
  v16h a;
#pragma unroll
  for (int i = 0; i < 8; ++i) { a[i] = p[i]; a[i + 8] = p[i + 16]; }
  return a;
}

// B tile 32x16 f16, element (k,n) at base[n*stride + k] (column-contiguous in k).
// Lanes 0-15: col n=lane, K 0..7 in b[0..7], K 16..23 in b[8..15]; lanes 16-31: K 8..15 / 24..31.
__device__ __forceinline__ v16h ld_b(const _Float16* base, int stride) {
  int lane = threadIdx.x & 31;
  int col  = lane & 15;
  int kb   = (lane >> 4) << 3;  // 0 or 8
  const _Float16* p = base + col * stride + kb;
  v16h b;
#pragma unroll
  for (int i = 0; i < 8; ++i) { b[i] = p[i]; b[i + 8] = p[i + 16]; }
  return b;
}

// ---------- CDNA5 async global->LDS (ASYNCcnt-tracked) -------------------
// GVS mode: LDS[vdst_lds] = MEM[saddr + voff]; 16B per lane.
__device__ __forceinline__ void async_ld_b128(unsigned lds_dst,
                                              const _Float16* base,
                                              unsigned byte_off) {
  asm volatile("global_load_async_to_lds_b128 %0, %1, %2"
               :: "v"(lds_dst), "v"(byte_off), "s"(base)
               : "memory");
}

// ---------- Phase 0: pack weights/bias/x to f16 --------------------------
__global__ void pack_kernel(const float* __restrict__ x,
                            const float* __restrict__ Wq, const float* __restrict__ bq,
                            const float* __restrict__ Wk, const float* __restrict__ bk,
                            const float* __restrict__ Wv, const float* __restrict__ bv,
                            _Float16* __restrict__ Wh, float* __restrict__ bias_all,
                            _Float16* __restrict__ xh) {
  int idx = blockIdx.x * blockDim.x + threadIdx.x;
  if (idx < kD * kC) {
    int d = idx / kC, c = idx % kC;
    float v = (d < kCQ)     ? Wq[d * kC + c]
            : (d < 2 * kCQ) ? Wk[(d - kCQ) * kC + c]
                            : Wv[(d - 2 * kCQ) * kC + c];
    Wh[idx] = (_Float16)v;
  }
  if (idx < kD) {
    bias_all[idx] = (idx < kCQ)     ? bq[idx]
                  : (idx < 2 * kCQ) ? bk[idx - kCQ]
                                    : bv[idx - 2 * kCQ];
  }
  size_t total = (size_t)kB * kC * kN;
  for (size_t i = idx; i < total; i += (size_t)gridDim.x * blockDim.x)
    xh[i] = (_Float16)x[i];
}

// ---------- Phase 1: fused QKV projection GEMM ----------------------------
// grid = (N/16, B), block = 256 (8 waves). Each wave: 5 of the 40 row-tiles.
__global__ __launch_bounds__(256) void proj_kernel(
    const _Float16* __restrict__ Wh, const float* __restrict__ bias_all,
    const _Float16* __restrict__ xh, _Float16* __restrict__ QT,
    _Float16* __restrict__ KT, _Float16* __restrict__ Vh) {
  constexpr int XS = kC + 8;  // padded stride (halves) against bank conflicts
  __shared__ _Float16 xT[16 * XS];  // x tile transposed: [n][c]

  int b = blockIdx.y, n0 = blockIdx.x * 16;
  int tid = threadIdx.x;
  const _Float16* xb = xh + (size_t)b * kC * kN;
  for (int t = tid; t < 16 * kC; t += 256) {
    int n = t & 15, c = t >> 4;
    xT[n * XS + c] = xb[(size_t)c * kN + n0 + n];
  }
  __syncthreads();

  int wave = tid >> 5, lane = tid & 31;
  int nl = lane & 15, hi = lane >> 4;
  for (int q = 0; q < 5; ++q) {
    int m0 = (wave * 5 + q) * 16;
    v8f acc = {};
    for (int ks = 0; ks < kC / 32; ++ks) {
      v16h a  = ld_a(Wh + (size_t)m0 * kC + ks * 32, kC);
      v16h bb = ld_b(xT + ks * 32, XS);
      acc = wmma_f16(a, bb, acc);
    }
    int n = n0 + nl;
#pragma unroll
    for (int r = 0; r < 8; ++r) {
      int m = m0 + r + hi * 8;
      float v = acc[r] + bias_all[m];
      if (m0 < kCQ)
        QT[((size_t)b * kN + n) * kCQ + m] = (_Float16)v;
      else if (m0 < 2 * kCQ)
        KT[((size_t)b * kN + n) * kCQ + (m - kCQ)] = (_Float16)v;
      else
        Vh[(size_t)b * kC * kN + (size_t)(m - 2 * kCQ) * kN + n] = (_Float16)v;
    }
  }
}

// ---------- Phase 2: flash attention + residual ---------------------------
// grid = (N/64, B), block = 256 (8 waves). Wave w owns output channels
// [64w, 64w+64); accumulators: 4x4 grid of 16x16 f32 WMMA tiles per wave.
// K tiles double-buffered via global_load_async_to_lds_b128 (ASYNCcnt).
__global__ __launch_bounds__(256) void attn_kernel(
    const _Float16* __restrict__ QT, const _Float16* __restrict__ KT,
    const _Float16* __restrict__ Vh, const float* __restrict__ x,
    const float* __restrict__ alpha, float* __restrict__ out) {
  constexpr int NJ = kN / 64;
  constexpr int QS = 72, KS = 72, PS = 72, SS = 65;  // padded LDS strides
  __shared__ __align__(16) _Float16 Qs[64 * QS];      // Q^T block: [i][d]
  __shared__ __align__(16) _Float16 Ksh[2][64 * KS];  // K^T blocks: [j][d]
  __shared__ __align__(16) _Float16 Psh[64 * PS];     // probs:      [i][j]
  __shared__ float    Ss[64 * SS];                    // logits:     [i][j]
  __shared__ float    m_s[64], l_s[64], corr_s[64];

  int b  = blockIdx.y;
  int i0 = blockIdx.x * 64;
  int tid = threadIdx.x, wave = tid >> 5, lane = tid & 31;
  int nl = lane & 15, hi = lane >> 4;

  const _Float16* QTb = QT + (size_t)b * kN * kCQ;
  const _Float16* KTb = KT + (size_t)b * kN * kCQ;
  const _Float16* Vb  = Vh + (size_t)b * kC * kN;

  // Chunk geometry: tile = 64 rows x 64 halves = 512 x 16B chunks; each
  // thread owns chunks tid and tid+256. Global side is linear (n-major
  // storage); LDS side uses the padded 144B row stride.
  unsigned l0 = ((unsigned)(tid >> 3) * KS + (unsigned)(tid & 7) * 8) * 2;
  unsigned l1 = ((unsigned)((tid + 256) >> 3) * KS + (unsigned)((tid + 256) & 7) * 8) * 2;
  unsigned g0 = (unsigned)tid * 16, g1 = (unsigned)(tid + 256) * 16;
  unsigned qsb = (unsigned)(uintptr_t)&Qs[0];
  unsigned ksb[2] = {(unsigned)(uintptr_t)&Ksh[0][0],
                     (unsigned)(uintptr_t)&Ksh[1][0]};

  // Async-stage Q block and K block 0.
  async_ld_b128(qsb + l0, QTb + (size_t)i0 * kCQ, g0);
  async_ld_b128(qsb + l1, QTb + (size_t)i0 * kCQ, g1);
  async_ld_b128(ksb[0] + l0, KTb, g0);
  async_ld_b128(ksb[0] + l1, KTb, g1);

  if (tid < 64) { m_s[tid] = -1e30f; l_s[tid] = 0.f; }

  v8f zero = {};
  v8f acc[4][4];
#pragma unroll
  for (int ct = 0; ct < 4; ++ct)
#pragma unroll
    for (int it = 0; it < 4; ++it) acc[ct][it] = zero;

  for (int jb = 0; jb < NJ; ++jb) {
    int j0 = jb * 64;
    int cur = jb & 1;
    const _Float16* Kcur = Ksh[cur];
    if (jb + 1 < NJ) {
      // prefetch next K tile into the other buffer, then wait for current
      const _Float16* kn = KTb + (size_t)(jb + 1) * 64 * kCQ;
      async_ld_b128(ksb[cur ^ 1] + l0, kn, g0);
      async_ld_b128(ksb[cur ^ 1] + l1, kn, g1);
      asm volatile("s_wait_asynccnt 0x2" ::: "memory");
    } else {
      asm volatile("s_wait_asynccnt 0x0" ::: "memory");
    }
    __syncthreads();

    // S(64x64) = Q^T K : 16 tiles over 8 waves (2 each)
    {
      int tj = wave & 3;
#pragma unroll
      for (int p = 0; p < 2; ++p) {
        int ti = ((wave >> 2) << 1) + p;
        v8f s = zero;
#pragma unroll
        for (int ks = 0; ks < 2; ++ks) {
          v16h a  = ld_a(Qs + ti * 16 * QS + ks * 32, QS);
          v16h bb = ld_b(Kcur + tj * 16 * KS + ks * 32, KS);
          s = wmma_f16(a, bb, s);
        }
#pragma unroll
        for (int r = 0; r < 8; ++r)
          Ss[(ti * 16 + r + hi * 8) * SS + tj * 16 + nl] = s[r];
      }
    }
    __syncthreads();

    // online softmax: 4 lanes per row, shfl_xor reductions within a wave
    {
      int row = tid >> 2, part = tid & 3;
      const float* srow = Ss + row * SS + part * 16;
      float mx = -1e30f;
#pragma unroll
      for (int c2 = 0; c2 < 16; ++c2) mx = fmaxf(mx, srow[c2]);
      mx = fmaxf(mx, __shfl_xor(mx, 1, 32));
      mx = fmaxf(mx, __shfl_xor(mx, 2, 32));
      float m_old = m_s[row];
      float m_new = fmaxf(m_old, mx);
      float lsum = 0.f;
      _Float16* prow = Psh + row * PS + part * 16;
#pragma unroll
      for (int c2 = 0; c2 < 16; ++c2) {
        float pv = __expf(srow[c2] - m_new);
        prow[c2] = (_Float16)pv;
        lsum += pv;
      }
      lsum += __shfl_xor(lsum, 1, 32);
      lsum += __shfl_xor(lsum, 2, 32);
      if (part == 0) {
        float cf = __expf(m_old - m_new);
        corr_s[row] = cf;
        l_s[row] = l_s[row] * cf + lsum;
        m_s[row] = m_new;
      }
    }
    __syncthreads();

    // rescale accumulators (factor depends only on i = n-dim of D tiles)
#pragma unroll
    for (int it = 0; it < 4; ++it) {
      float cf = corr_s[it * 16 + nl];
#pragma unroll
      for (int ct = 0; ct < 4; ++ct)
#pragma unroll
        for (int r = 0; r < 8; ++r) acc[ct][it][r] *= cf;
    }
    // O += V * P^T : D[m=c][n=i] = sum_j V[c][j] * P[i][j]
#pragma unroll
    for (int ct = 0; ct < 4; ++ct) {
      int c0 = wave * 64 + ct * 16;
      if (jb + 1 < NJ)
        __builtin_prefetch(Vb + (size_t)c0 * kN + j0 + 64, 0, 0);
#pragma unroll
      for (int ks = 0; ks < 2; ++ks) {
        v16h a = ld_a(Vb + (size_t)c0 * kN + j0 + ks * 32, kN);
#pragma unroll
        for (int it = 0; it < 4; ++it) {
          v16h bb = ld_b(Psh + it * 16 * PS + ks * 32, PS);
          acc[ct][it] = wmma_f16(a, bb, acc[ct][it]);
        }
      }
    }
    __syncthreads();
  }

  float al = alpha[0];
#pragma unroll
  for (int it = 0; it < 4; ++it) {
    float linv = 1.0f / l_s[it * 16 + nl];
    int i = i0 + it * 16 + nl;
#pragma unroll
    for (int ct = 0; ct < 4; ++ct) {
#pragma unroll
      for (int r = 0; r < 8; ++r) {
        int c = wave * 64 + ct * 16 + r + hi * 8;
        size_t idx = ((size_t)b * kC + c) * (size_t)kN + i;
        out[idx] = al * acc[ct][it][r] * linv + x[idx];
      }
    }
  }
}

extern "C" void kernel_launch(void* const* d_in, const int* in_sizes, int n_in,
                              void* d_out, int out_size, void* d_ws, size_t ws_size,
                              hipStream_t stream) {
  const float* x     = (const float*)d_in[0];
  const float* Wq    = (const float*)d_in[1];
  const float* bq    = (const float*)d_in[2];
  const float* Wk    = (const float*)d_in[3];
  const float* bk    = (const float*)d_in[4];
  const float* Wv    = (const float*)d_in[5];
  const float* bv    = (const float*)d_in[6];
  const float* alpha = (const float*)d_in[7];
  float* out = (float*)d_out;

  char* ws = (char*)d_ws;
  _Float16* Wh       = (_Float16*)(ws + WH_OFF);
  float*    bias_all = (float*)(ws + BIAS_OFF);
  _Float16* xh       = (_Float16*)(ws + XH_OFF);
  _Float16* QT       = (_Float16*)(ws + QT_OFF);
  _Float16* KT       = (_Float16*)(ws + KT_OFF);
  _Float16* Vh       = (_Float16*)(ws + VH_OFF);

  pack_kernel<<<4096, 256, 0, stream>>>(x, Wq, bq, Wk, bk, Wv, bv, Wh, bias_all, xh);
  proj_kernel<<<dim3(kN / 16, kB), 256, 0, stream>>>(Wh, bias_all, xh, QT, KT, Vh);
  attn_kernel<<<dim3(kN / 64, kB), 256, 0, stream>>>(QT, KT, Vh, x, alpha, out);
}
